// TweetClassifier_34265249088416
// MI455X (gfx1250) — matrix operand order repaired and checked by reading, not compiled, not run on hardware
//
#include <hip/hip_runtime.h>
#include <hip/hip_bf16.h>
#include <stdint.h>

typedef _Float16 v8h_t  __attribute__((ext_vector_type(8)));
typedef _Float16 v16h_t __attribute__((ext_vector_type(16)));
typedef float    v8f_t  __attribute__((ext_vector_type(8)));

constexpr int BB = 32;   // batch
constexpr int TT = 64;   // tweets per batch
constexpr int WW = 64;   // words per tweet
constexpr int DD = 512;  // word dim
constexpr int HH = 512;  // hidden
constexpr int G4 = 4 * HH;        // 2048 gate width
constexpr int KC = DD + HH;       // 1024 concat K
constexpr unsigned NBLK = 64;     // persistent-kernel grid (512 waves)
constexpr int ROWSP = KC + 8;     // padded LDS row stride (halves): 516 DW = 4 mod 64 banks
constexpr size_t LDS_BYTES = (size_t)4 * 16 * ROWSP * sizeof(_Float16); // 132096

// ---------------------------------------------------------------------------
// Device-scope sense-reversing grid barrier (64 co-resident blocks).
// ---------------------------------------------------------------------------
__device__ __forceinline__ void grid_barrier(unsigned* cnt, unsigned* gen)
{
    __threadfence();
    __syncthreads();
    if (threadIdx.x == 0) {
        unsigned g = __hip_atomic_load(gen, __ATOMIC_RELAXED,
                                       __HIP_MEMORY_SCOPE_AGENT);
        unsigned arrived = __hip_atomic_fetch_add(cnt, 1u, __ATOMIC_ACQ_REL,
                                                  __HIP_MEMORY_SCOPE_AGENT);
        if (arrived == NBLK - 1u) {
            __hip_atomic_store(cnt, 0u, __ATOMIC_RELAXED,
                               __HIP_MEMORY_SCOPE_AGENT);
            __hip_atomic_fetch_add(gen, 1u, __ATOMIC_RELEASE,
                                   __HIP_MEMORY_SCOPE_AGENT);
        } else {
            while (__hip_atomic_load(gen, __ATOMIC_ACQUIRE,
                                     __HIP_MEMORY_SCOPE_AGENT) == g)
                __builtin_amdgcn_s_sleep(1);
        }
    }
    __syncthreads();
}

// ---------------------------------------------------------------------------
// Kernel 1: word attention -> masked softmax -> pooled tweet embedding (f16)
// ---------------------------------------------------------------------------
__global__ void attention_kernel(const float* __restrict__ x,
                                 const int* __restrict__ n_tweets,
                                 const int* __restrict__ n_words,
                                 const float* __restrict__ w_att,
                                 const float* __restrict__ b_att,
                                 _Float16* __restrict__ x16)
{
    int t = blockIdx.x, b = blockIdx.y;
    const float* xa = x + (((size_t)b * TT + t) * WW) * DD;
    __shared__ float sc[WW];
    __shared__ float at[WW];
    int tid = threadIdx.x, lane = tid & 31, wv = tid >> 5;
    int nw = n_words[b * TT + t];
    float batt = *b_att;

    for (int w = wv; w < WW; w += 8) {
        const float* xw = xa + (size_t)w * DD;
        float p = 0.f;
        for (int d = lane; d < DD; d += 32) p += xw[d] * w_att[d];
        for (int off = 16; off; off >>= 1) p += __shfl_xor(p, off, 32);
        if (lane == 0) sc[w] = (w < nw) ? (p + batt) : -1e30f;
    }
    __syncthreads();

    if (tid < WW) {
        float mx = -3.4e38f;
        for (int w = 0; w < WW; ++w) mx = fmaxf(mx, sc[w]);
        float sum = 0.f;
        for (int w = 0; w < WW; ++w) sum += __expf(sc[w] - mx);
        at[tid] = __expf(sc[tid] - mx) / sum;
    }
    __syncthreads();

    bool tmask = t < n_tweets[b];
    for (int d = tid; d < DD; d += 256) {
        float acc = 0.f;
        if (tmask) {
            for (int w = 0; w < WW; ++w) acc += at[w] * xa[(size_t)w * DD + d];
        }
        x16[((size_t)b * TT + t) * DD + d] = (_Float16)acc;
    }
}

// ---------------------------------------------------------------------------
// Kernel 2: pack [Whh | Wih] per direction into row-major f16 [2][2048][1024]
// ---------------------------------------------------------------------------
__global__ void pack_weights(const float* __restrict__ Wih_f,
                             const float* __restrict__ Whh_f,
                             const float* __restrict__ Wih_b,
                             const float* __restrict__ Whh_b,
                             _Float16* __restrict__ wcat)
{
    size_t total = (size_t)2 * G4 * KC;
    for (size_t i = (size_t)blockIdx.x * blockDim.x + threadIdx.x; i < total;
         i += (size_t)gridDim.x * blockDim.x) {
        int dir = (int)(i >> 21);
        size_t rem = i & ((1u << 21) - 1);
        int n = (int)(rem >> 10);
        int k = (int)(rem & 1023);
        const float* Whh = dir ? Whh_b : Whh_f;
        const float* Wih = dir ? Wih_b : Wih_f;
        float v = (k < HH) ? Whh[(size_t)n * HH + k]
                           : Wih[(size_t)n * DD + (k - HH)];
        wcat[i] = (_Float16)v;
    }
}

// ---------------------------------------------------------------------------
// Kernel 3: zero barrier words + precombine biases  bsum[dir][n] = bih+bhh
// ---------------------------------------------------------------------------
__global__ void misc_init(const float* __restrict__ bih_f,
                          const float* __restrict__ bhh_f,
                          const float* __restrict__ bih_b,
                          const float* __restrict__ bhh_b,
                          float* __restrict__ bsum, unsigned* bar)
{
    int i = blockIdx.x * blockDim.x + threadIdx.x;
    if (i < 2) bar[i] = 0u;
    if (i < 2 * G4) {
        int d = i >> 11;
        int n = i & (G4 - 1);
        bsum[i] = d ? (bih_b[n] + bhh_b[n]) : (bih_f[n] + bhh_f[n]);
    }
}

// ---------------------------------------------------------------------------
// Kernel 4: persistent bidirectional LSTM scan, weights staged in LDS.
// Block b: dir = b>>5, ntiles = (b&31)*4 .. +3, waves (0..7): ntile=wv>>1,
// mtile=wv&1 (mtile pair shares B rows -> 128 KB LDS weights per block).
// Per step: WMMA GEMM phase, grid barrier, gate update phase, grid barrier.
// ---------------------------------------------------------------------------
__global__ void lstm_persistent(const _Float16* __restrict__ x16,
                                const _Float16* __restrict__ wcat,
                                float* __restrict__ z,
                                float* __restrict__ h,
                                float* __restrict__ c,
                                _Float16* __restrict__ h16,
                                const float* __restrict__ bsum,
                                const int* __restrict__ n_tweets,
                                unsigned* bar)
{
    extern __shared__ _Float16 lws[];   // [4 tiles][16 rows][ROWSP halves]
    unsigned* cnt = bar;
    unsigned* gen = bar + 1;
    int gtid = (int)(blockIdx.x * blockDim.x + threadIdx.x);   // 0..16383

    int dir = (int)blockIdx.x >> 5;
    int base_nt = ((int)blockIdx.x & 31) * 4;

    // --- stage this block's 64 weight rows (4 n-tiles) into LDS ---
    for (int ch = threadIdx.x; ch < 64 * 128; ch += 256) {
        int row = ch >> 7;          // 0..63
        int c8 = ch & 127;          // 8-half chunks along K
        const v8h_t* src = (const v8h_t*)(wcat +
            ((size_t)dir * G4 + base_nt * 16 + row) * KC + c8 * 8);
        *(v8h_t*)(&lws[row * ROWSP + c8 * 8]) = *src;
    }

    // --- init h/c/h16 to zero ---
    for (int e = gtid; e < 2 * BB * HH; e += (int)(NBLK * 256)) {
        h[e] = 0.f;
        c[e] = 0.f;
        h16[e] = (_Float16)0.f;
    }
    grid_barrier(cnt, gen);   // also covers the LDS fill (__syncthreads inside)

    // --- constant wave->tile mapping ---
    int lane = threadIdx.x & 31;
    int wv = (int)(threadIdx.x >> 5);        // 0..7
    int ntile = base_nt + (wv >> 1);         // global N tile
    int mtile = wv & 1;
    int lm = lane & 15;
    int hi = (lane >> 4) & 1;
    int m = mtile * 16 + lm;                 // batch row 0..31
    const _Float16* hrow = h16 + ((size_t)dir * BB + m) * HH;
    const _Float16* lbase = &lws[((wv >> 1) * 16 + lm) * ROWSP + hi * 8];
    int aoff = hi * 8;
    float* zr = z + (size_t)dir * BB * G4;
    int mbase = mtile * 16 + hi * 8;

    for (int t = 0; t < TT; ++t) {
        // ---------------- GEMM phase ----------------
        int t_eff = dir ? (TT - 1 - t) : t;
        const _Float16* xrow = x16 + ((size_t)m * TT + t_eff) * DD;

        v8f_t acc = {};
#pragma unroll
        for (int kb = 0; kb < KC; kb += 32) {
            // A fragment (global): h for k<512, x_t for k>=512
            const _Float16* abase = (kb < HH) ? (hrow + kb) : (xrow + (kb - HH));
            v8h_t alo = *(const v8h_t*)(abase + aoff);
            v8h_t ahi = *(const v8h_t*)(abase + aoff + 16);
            v16h_t av = __builtin_shufflevector(alo, ahi, 0, 1, 2, 3, 4, 5, 6,
                                                7, 8, 9, 10, 11, 12, 13, 14, 15);
            // B fragment (LDS): lane's column rows, k = 8h..8h+7 and 16+8h..16+8h+7
            v8h_t blo = *(const v8h_t*)(lbase + kb);
            v8h_t bhi = *(const v8h_t*)(lbase + kb + 16);
            v16h_t bv = __builtin_shufflevector(blo, bhi, 0, 1, 2, 3, 4, 5, 6,
                                                7, 8, 9, 10, 11, 12, 13, 14, 15);
            acc = __builtin_amdgcn_wmma_f32_16x16x32_f16(
                false, av, false, bv, (short)0, acc, false, false);
        }
        // C/D layout: lane<16 -> M=v, N=lane ; lane>=16 -> M=v+8, N=lane-16
#pragma unroll
        for (int v = 0; v < 8; ++v)
            zr[(size_t)(mbase + v) * G4 + ntile * 16 + lm] = acc[v];

        grid_barrier(cnt, gen);        // z visible everywhere

        // ---------------- update phase ----------------
        for (int e = gtid; e < 2 * BB * HH; e += (int)(NBLK * 256)) {
            int ed = e >> 14;
            int er = e & 16383;
            int b = er >> 9;
            int j = er & 511;
            const float* zq = z + (size_t)(ed * BB + b) * G4;
            const float* bs = bsum + (size_t)ed * G4;

            float iv = zq[j]          + bs[j];
            float fv = zq[HH + j]     + bs[HH + j];
            float gv = zq[2 * HH + j] + bs[2 * HH + j];
            float ov = zq[3 * HH + j] + bs[3 * HH + j];

            int te = ed ? (TT - 1 - t) : t;
            bool msk = te < n_tweets[b];

            size_t ci = (size_t)(ed * BB + b) * HH + j;
            float co = c[ci], ho = h[ci];
            float si = 1.f / (1.f + __expf(-iv));
            float sf = 1.f / (1.f + __expf(-fv));
            float so = 1.f / (1.f + __expf(-ov));
            float cn = sf * co + si * tanhf(gv);
            float hn = so * tanhf(cn);
            float cw = msk ? cn : co;
            float hw = msk ? hn : ho;
            c[ci] = cw;
            h[ci] = hw;
            h16[ci] = (_Float16)hw;
        }

        grid_barrier(cnt, gen);        // new h16 visible before next step
    }
}

// ---------------------------------------------------------------------------
// Kernel 5: classifier  sigmoid([h_f | h_b] . w_cls + b_cls)
// ---------------------------------------------------------------------------
__global__ void final_kernel(const float* __restrict__ h,
                             const float* __restrict__ w_cls,
                             const float* __restrict__ b_cls,
                             float* __restrict__ out)
{
    int b = blockIdx.x;
    int tid = threadIdx.x;
    __shared__ float red[256];
    float p = 0.f;
    for (int k = tid; k < 2 * HH; k += 256) {
        float hv = (k < HH) ? h[(size_t)b * HH + k]
                            : h[(size_t)(BB + b) * HH + (k - HH)];
        p += hv * w_cls[k];
    }
    red[tid] = p;
    __syncthreads();
    for (int s = 128; s; s >>= 1) {
        if (tid < s) red[tid] += red[tid + s];
        __syncthreads();
    }
    if (tid == 0) out[b] = 1.f / (1.f + __expf(-(red[0] + *b_cls)));
}

// ---------------------------------------------------------------------------
extern "C" void kernel_launch(void* const* d_in, const int* in_sizes, int n_in,
                              void* d_out, int out_size, void* d_ws, size_t ws_size,
                              hipStream_t stream)
{
    const float* input_ids = (const float*)d_in[0];
    const int*   n_tweets  = (const int*)d_in[1];
    const int*   n_words   = (const int*)d_in[2];
    const float* w_att     = (const float*)d_in[3];
    const float* b_att     = (const float*)d_in[4];
    const float* Wih_f     = (const float*)d_in[5];
    const float* Whh_f     = (const float*)d_in[6];
    const float* bih_f     = (const float*)d_in[7];
    const float* bhh_f     = (const float*)d_in[8];
    const float* Wih_b     = (const float*)d_in[9];
    const float* Whh_b     = (const float*)d_in[10];
    const float* bih_b     = (const float*)d_in[11];
    const float* bhh_b     = (const float*)d_in[12];
    const float* w_cls     = (const float*)d_in[13];
    const float* b_cls     = (const float*)d_in[14];
    float* out = (float*)d_out;

    // workspace carve-out
    uint8_t* ws = (uint8_t*)d_ws;
    _Float16* x16  = (_Float16*)ws;                       // 2 MB
    size_t off = (size_t)BB * TT * DD * 2;
    _Float16* wcat = (_Float16*)(ws + off);               // 8 MB
    off += (size_t)2 * G4 * KC * 2;
    float* zbuf = (float*)(ws + off);                     // 512 KB
    off += (size_t)2 * BB * G4 * 4;
    float* hbuf = (float*)(ws + off);                     // 128 KB
    off += (size_t)2 * BB * HH * 4;
    float* cbuf = (float*)(ws + off);                     // 128 KB
    off += (size_t)2 * BB * HH * 4;
    _Float16* h16 = (_Float16*)(ws + off);                // 64 KB
    off += (size_t)2 * BB * HH * 2;
    float* bsum = (float*)(ws + off);                     // 16 KB
    off += (size_t)2 * G4 * 4;
    unsigned* bar = (unsigned*)(ws + off);                // 2 words

    // allow 132 KB dynamic LDS (ignore error; idempotent, capture-safe)
    (void)hipFuncSetAttribute((const void*)lstm_persistent,
                              hipFuncAttributeMaxDynamicSharedMemorySize,
                              (int)LDS_BYTES);

    attention_kernel<<<dim3(TT, BB), 256, 0, stream>>>(
        input_ids, n_tweets, n_words, w_att, b_att, x16);

    pack_weights<<<2048, 256, 0, stream>>>(Wih_f, Whh_f, Wih_b, Whh_b, wcat);

    misc_init<<<(2 * G4 + 255) / 256, 256, 0, stream>>>(
        bih_f, bhh_f, bih_b, bhh_b, bsum, bar);

    lstm_persistent<<<NBLK, 256, LDS_BYTES, stream>>>(
        x16, wcat, zbuf, hbuf, cbuf, h16, bsum, n_tweets, bar);

    final_kernel<<<BB, 256, 0, stream>>>(hbuf, w_cls, b_cls, out);
}
